// AttnBlock_3650722202286
// MI455X (gfx1250) — matrix operand order repaired and checked by reading, not compiled, not run on hardware
//
#include <hip/hip_runtime.h>
#include <hip/hip_bf16.h>
#include <stdint.h>

typedef _Float16 h16;
typedef __attribute__((ext_vector_type(8)))  _Float16 v8h;
typedef __attribute__((ext_vector_type(16))) _Float16 v16h;
typedef __attribute__((ext_vector_type(8)))  float    v8f;

#define BATCH 4
#define CCH   512
#define HW    4096
#define NGRP  32
#define CPG   16
#define GN_EPS 1e-6f

// ---------------------------------------------------------------------------
// CDNA5 async global->LDS copy (16 bytes per lane), ASYNCcnt tracked.
// ---------------------------------------------------------------------------
__device__ __forceinline__ void async_copy16(void* lds_dst, const void* gsrc) {
    unsigned lds = (unsigned)(uintptr_t)lds_dst;   // low 32 bits = LDS byte offset
    asm volatile("global_load_async_to_lds_b128 %0, %1, off"
                 :: "v"(lds), "v"(gsrc) : "memory");
}
__device__ __forceinline__ void async_wait0() {
    asm volatile("s_wait_asynccnt 0x0" ::: "memory");
}
__device__ __forceinline__ void async_wait4() {
    asm volatile("s_wait_asynccnt 0x4" ::: "memory");
}

// ---------------------------------------------------------------------------
// GroupNorm statistics: one block per (batch, group).
// ---------------------------------------------------------------------------
__global__ __launch_bounds__(256) void gn_stats(const float* __restrict__ x,
                                                float* __restrict__ mr) {
    int bg = blockIdx.x;
    int b = bg >> 5, g = bg & 31;
    const float* p = x + ((size_t)b * CCH + (size_t)g * CPG) * HW;
    float s = 0.f, s2 = 0.f;
    for (int i = threadIdx.x; i < CPG * HW; i += 256) {
        float v = p[i];
        s += v; s2 += v * v;
    }
    __shared__ float sh[2][256];
    sh[0][threadIdx.x] = s; sh[1][threadIdx.x] = s2;
    __syncthreads();
    for (int w = 128; w > 0; w >>= 1) {
        if (threadIdx.x < w) {
            sh[0][threadIdx.x] += sh[0][threadIdx.x + w];
            sh[1][threadIdx.x] += sh[1][threadIdx.x + w];
        }
        __syncthreads();
    }
    if (threadIdx.x == 0) {
        const float inv_n = 1.0f / (CPG * HW);
        float m   = sh[0][0] * inv_n;
        float var = sh[1][0] * inv_n - m * m;
        mr[bg * 2 + 0] = m;
        mr[bg * 2 + 1] = rsqrtf(var + GN_EPS);
    }
}

// ---------------------------------------------------------------------------
// Normalize + affine; emit f16 in (hw, c) layout only (all GEMMs are NT-form).
// ---------------------------------------------------------------------------
__global__ __launch_bounds__(256) void gn_norm(const float* __restrict__ x,
                                               const float* __restrict__ gw,
                                               const float* __restrict__ gb,
                                               const float* __restrict__ mr,
                                               h16* __restrict__ xnT) {
    int bc = blockIdx.x;
    int b = bc >> 9, c = bc & 511;
    int g = c >> 4;
    float m = mr[(b * NGRP + g) * 2 + 0];
    float r = mr[(b * NGRP + g) * 2 + 1];
    float ga = gw[c] * r;
    float be = gb[c] - m * ga;
    const float* px = x + (size_t)bc * HW;
    h16* pt = xnT + (size_t)b * HW * CCH + c;
    for (int i = threadIdx.x; i < HW; i += 256)
        pt[(size_t)i * CCH] = (h16)(px[i] * ga + be);
}

// ---------------------------------------------------------------------------
// Weight prep: straight f32 -> f16 (no transposes needed in NT-form).
// ---------------------------------------------------------------------------
__global__ __launch_bounds__(256) void prep_w(const float* __restrict__ wq,
                                              const float* __restrict__ wk,
                                              const float* __restrict__ wv,
                                              const float* __restrict__ wo,
                                              h16* __restrict__ wqh,
                                              h16* __restrict__ wkh,
                                              h16* __restrict__ wvh,
                                              h16* __restrict__ woh) {
    int i = blockIdx.x * 256 + threadIdx.x;
    wqh[i] = (h16)wq[i];
    wkh[i] = (h16)wk[i];
    wvh[i] = (h16)wv[i];
    woh[i] = (h16)wo[i];
}

// ---------------------------------------------------------------------------
// NT-form 128x128 WMMA f16 GEMM, f32 accumulate, async-LDS double buffering.
//   A : M x K row-major (lda)      -- K contiguous
//   Bt: N x K row-major (ldb)      -- K contiguous (i.e. B transposed)
//   C[m][n] = sum_k A[m][k] * Bt[n][k]
//   mode 0: f16 store, (acc + bias[col]) * scale   (bias may be null)
//   mode 1: f16 store,  acc + bias[row]
//   mode 2: f32 store, raw
//   mode 3: f32 transposed store + bias[col] + residual
// ---------------------------------------------------------------------------
#define BM 128
#define BN 128
#define BK 32

__global__ __launch_bounds__(256) void gemm_wmma(
    const h16* __restrict__ A, int lda, long long strideA,
    const h16* __restrict__ Bt, int ldb, long long strideB,
    void* __restrict__ Cv, int ldc, long long strideC,
    int K,
    const float* __restrict__ bias, float scale, int mode,
    const float* __restrict__ resid, long long strideR)
{
    __shared__ __align__(16) h16 As[2][BM][BK + 8];
    __shared__ __align__(16) h16 Bs[2][BN][BK + 8];

    const int tid  = threadIdx.x;
    const int lane = tid & 31;
    const int wid  = tid >> 5;            // 8 waves: 2 (M) x 4 (N)
    const int wm   = wid >> 2;
    const int wn   = wid & 3;
    const int l15  = lane & 15;
    const int bm   = blockIdx.y * BM;
    const int bn   = blockIdx.x * BN;
    const int bz   = blockIdx.z;

    A  += (size_t)bz * strideA;
    Bt += (size_t)bz * strideB;

    // staging slice for this thread: row r (of 128), halves [c0, c0+16)
    const int sr = tid >> 1;
    const int sc = (tid & 1) * 16;

    v8f acc[4][2];
#pragma unroll
    for (int mt = 0; mt < 4; ++mt)
#pragma unroll
        for (int nt = 0; nt < 2; ++nt)
#pragma unroll
            for (int e = 0; e < 8; ++e) acc[mt][nt][e] = 0.f;

    const int kbA = (lane < 16) ? 0 : 8;   // A-frag K base (ISA 7.12.2)
    const int kbB = (lane < 16) ? 0 : 8;   // B-frag K base

    // prologue: stage tile 0 into buffer 0
    {
        const h16* ga = A  + (size_t)(bm + sr) * lda + sc;
        const h16* gb = Bt + (size_t)(bn + sr) * ldb + sc;
        async_copy16(&As[0][sr][sc],     ga);
        async_copy16(&As[0][sr][sc + 8], ga + 8);
        async_copy16(&Bs[0][sr][sc],     gb);
        async_copy16(&Bs[0][sr][sc + 8], gb + 8);
    }

    int buf = 0;
    for (int k0 = 0; k0 < K; k0 += BK) {
        const bool has_next = (k0 + BK) < K;
        if (has_next) {   // stage tile k0+BK into the other buffer
            const h16* ga = A  + (size_t)(bm + sr) * lda + (k0 + BK) + sc;
            const h16* gb = Bt + (size_t)(bn + sr) * ldb + (k0 + BK) + sc;
            async_copy16(&As[buf ^ 1][sr][sc],     ga);
            async_copy16(&As[buf ^ 1][sr][sc + 8], ga + 8);
            async_copy16(&Bs[buf ^ 1][sr][sc],     gb);
            async_copy16(&Bs[buf ^ 1][sr][sc + 8], gb + 8);
            async_wait4();          // tile k0's 4 copies done, next 4 in flight
        } else {
            async_wait0();
        }
        __syncthreads();

        // fragments: both operands are 2 x ds_load_b128 per subtile
        v16h af[4];
#pragma unroll
        for (int mt = 0; mt < 4; ++mt) {
            const h16* ap = &As[buf][wm * 64 + mt * 16 + l15][0];
            v8h lo = *(const v8h*)(ap + kbA);        // K = kb   .. kb+7
            v8h hi = *(const v8h*)(ap + kbA + 16);   // K = kb+16.. kb+23
#pragma unroll
            for (int e = 0; e < 8; ++e) { af[mt][e] = lo[e]; af[mt][8 + e] = hi[e]; }
        }
        v16h bf[2];
#pragma unroll
        for (int nt = 0; nt < 2; ++nt) {
            const h16* bp = &Bs[buf][wn * 32 + nt * 16 + l15][0];
            v8h lo = *(const v8h*)(bp + kbB);        // K = kb .. kb+7
            v8h hi = *(const v8h*)(bp + kbB + 16);   // K = kb+16.. kb+23
#pragma unroll
            for (int e = 0; e < 8; ++e) { bf[nt][e] = lo[e]; bf[nt][8 + e] = hi[e]; }
        }

#pragma unroll
        for (int mt = 0; mt < 4; ++mt)
#pragma unroll
            for (int nt = 0; nt < 2; ++nt)
                acc[mt][nt] = __builtin_amdgcn_wmma_f32_16x16x32_f16(
                    false, af[mt], false, bf[nt], (short)0, acc[mt][nt],
                    false, false);

        __syncthreads();   // everyone done reading buf before it is re-staged
        buf ^= 1;
    }

    // --- epilogue ---
    const int mhalf = (lane < 16) ? 0 : 8;
    if (mode == 0) {
        h16* Ch = (h16*)Cv + (size_t)bz * strideC;
#pragma unroll
        for (int mt = 0; mt < 4; ++mt)
#pragma unroll
            for (int nt = 0; nt < 2; ++nt)
#pragma unroll
                for (int v = 0; v < 8; ++v) {
                    int gm = bm + wm * 64 + mt * 16 + v + mhalf;
                    int gn = bn + wn * 32 + nt * 16 + l15;
                    float bb = bias ? bias[gn] : 0.f;
                    Ch[(size_t)gm * ldc + gn] = (h16)((acc[mt][nt][v] + bb) * scale);
                }
    } else if (mode == 1) {
        h16* Ch = (h16*)Cv + (size_t)bz * strideC;
#pragma unroll
        for (int mt = 0; mt < 4; ++mt)
#pragma unroll
            for (int nt = 0; nt < 2; ++nt)
#pragma unroll
                for (int v = 0; v < 8; ++v) {
                    int gm = bm + wm * 64 + mt * 16 + v + mhalf;
                    int gn = bn + wn * 32 + nt * 16 + l15;
                    Ch[(size_t)gm * ldc + gn] = (h16)(acc[mt][nt][v] + bias[gm]);
                }
    } else if (mode == 2) {
        float* Cf = (float*)Cv;
#pragma unroll
        for (int mt = 0; mt < 4; ++mt)
#pragma unroll
            for (int nt = 0; nt < 2; ++nt)
#pragma unroll
                for (int v = 0; v < 8; ++v) {
                    int gm = bm + wm * 64 + mt * 16 + v + mhalf;
                    int gn = bn + wn * 32 + nt * 16 + l15;
                    Cf[(size_t)gm * ldc + gn] = acc[mt][nt][v];
                }
    } else { // mode 3
        float* Cf = (float*)Cv + (size_t)bz * strideC;
        const float* R = resid + (size_t)bz * strideR;
#pragma unroll
        for (int mt = 0; mt < 4; ++mt)
#pragma unroll
            for (int nt = 0; nt < 2; ++nt)
#pragma unroll
                for (int v = 0; v < 8; ++v) {
                    int gm = bm + wm * 64 + mt * 16 + v + mhalf;   // spatial i
                    int gn = bn + wn * 32 + nt * 16 + l15;         // channel o
                    size_t idx = (size_t)gn * ldc + gm;
                    Cf[idx] = acc[mt][nt][v] + bias[gn] + R[idx];
                }
    }
}

// ---------------------------------------------------------------------------
// Row softmax: f32 logits (row of 4096) -> f16 probabilities.
// ---------------------------------------------------------------------------
__global__ __launch_bounds__(256) void softmax_rows(const float* __restrict__ S,
                                                    h16* __restrict__ P) {
    const int tid = threadIdx.x;
    const float* p = S + (size_t)blockIdx.x * HW;
    h16* q = P + (size_t)blockIdx.x * HW;
    float r[16];
    float mx = -3.4e38f;
#pragma unroll
    for (int j = 0; j < 16; ++j) {
        r[j] = p[tid + j * 256];
        mx = fmaxf(mx, r[j]);
    }
    __shared__ float sh[256];
    sh[tid] = mx; __syncthreads();
    for (int w = 128; w > 0; w >>= 1) {
        if (tid < w) sh[tid] = fmaxf(sh[tid], sh[tid + w]);
        __syncthreads();
    }
    mx = sh[0];
    __syncthreads();
    float s = 0.f;
#pragma unroll
    for (int j = 0; j < 16; ++j) {
        r[j] = __expf(r[j] - mx);
        s += r[j];
    }
    sh[tid] = s; __syncthreads();
    for (int w = 128; w > 0; w >>= 1) {
        if (tid < w) sh[tid] += sh[tid + w];
        __syncthreads();
    }
    float inv = 1.0f / sh[0];
#pragma unroll
    for (int j = 0; j < 16; ++j)
        q[tid + j * 256] = (h16)(r[j] * inv);
}

// ---------------------------------------------------------------------------
extern "C" void kernel_launch(void* const* d_in, const int* in_sizes, int n_in,
                              void* d_out, int out_size, void* d_ws, size_t ws_size,
                              hipStream_t stream) {
    (void)in_sizes; (void)n_in; (void)out_size; (void)ws_size;
    const float* x   = (const float*)d_in[0];
    const float* gnw = (const float*)d_in[1];
    const float* gnb = (const float*)d_in[2];
    const float* wq  = (const float*)d_in[3];
    const float* bq  = (const float*)d_in[4];
    const float* wk  = (const float*)d_in[5];
    const float* bk  = (const float*)d_in[6];
    const float* wv  = (const float*)d_in[7];
    const float* bv  = (const float*)d_in[8];
    const float* wo  = (const float*)d_in[9];
    const float* bo  = (const float*)d_in[10];
    float* out = (float*)d_out;

    const size_t SZ_XN = (size_t)BATCH * CCH * HW * sizeof(h16);   // 16.78 MB
    const size_t SZ_W  = (size_t)CCH * CCH * sizeof(h16);          // 0.52 MB

    uint8_t* w = (uint8_t*)d_ws;
    size_t off = 0;
    float* mr  = (float*)(w + off); off += 1024;
    h16* xnT   = (h16*)(w + off);   off += SZ_XN;   // (b, hw, c)
    h16* wqh   = (h16*)(w + off);   off += SZ_W;
    h16* wkh   = (h16*)(w + off);   off += SZ_W;
    h16* wvh   = (h16*)(w + off);   off += SZ_W;
    h16* woh   = (h16*)(w + off);   off += SZ_W;
    h16* qt    = (h16*)(w + off);   off += SZ_XN;   // (b, hw, c)
    h16* kt    = (h16*)(w + off);   off += SZ_XN;   // (b, hw, c)
    h16* vh    = (h16*)(w + off);   off += SZ_XN;   // (b, c, hw)
    h16* ot    = (h16*)(w + off);   off += SZ_XN;   // (b, hw, c)
    float* Sf  = (float*)(w + off); off += (size_t)HW * HW * sizeof(float);
    h16* Ph    = (h16*)(w + off);   off += (size_t)HW * HW * sizeof(h16);

    const long long sBC = (long long)CCH * HW;
    const float scale = 0.044194173824159216f;   // 512^-0.5

    gn_stats<<<128, 256, 0, stream>>>(x, mr);
    gn_norm<<<BATCH * CCH, 256, 0, stream>>>(x, gnw, gnb, mr, xnT);
    prep_w<<<(CCH * CCH) / 256, 256, 0, stream>>>(wq, wk, wv, wo, wqh, wkh, wvh, woh);

    // Qt = (xnT @ wq^T + bq) * s   -> qt (hw,c)      M=4096 N=512 K=512
    gemm_wmma<<<dim3(CCH / BN, HW / BM, BATCH), 256, 0, stream>>>(
        xnT, CCH, sBC, wqh, CCH, 0, qt, CCH, sBC, CCH, bq, scale, 0, nullptr, 0);
    // Kt = xnT @ wk^T + bk         -> kt (hw,c)
    gemm_wmma<<<dim3(CCH / BN, HW / BM, BATCH), 256, 0, stream>>>(
        xnT, CCH, sBC, wkh, CCH, 0, kt, CCH, sBC, CCH, bk, 1.f, 0, nullptr, 0);
    // V  = wv @ xn + bv            -> vh (c,hw)      M=512 N=4096 K=512
    gemm_wmma<<<dim3(HW / BN, CCH / BM, BATCH), 256, 0, stream>>>(
        wvh, CCH, 0, xnT, CCH, sBC, vh, HW, sBC, CCH, bv, 1.f, 1, nullptr, 0);

    for (int b = 0; b < BATCH; ++b) {
        // S = Qt @ Kt^T (f32 logits)                  M=4096 N=4096 K=512
        gemm_wmma<<<dim3(HW / BN, HW / BM, 1), 256, 0, stream>>>(
            qt + (size_t)b * sBC, CCH, 0, kt + (size_t)b * sBC, CCH, 0,
            Sf, HW, 0, CCH, nullptr, 1.f, 2, nullptr, 0);
        softmax_rows<<<HW, 256, 0, stream>>>(Sf, Ph);
        // Ot = P @ V^T (f16)                          M=4096 N=512 K=4096
        gemm_wmma<<<dim3(CCH / BN, HW / BM, 1), 256, 0, stream>>>(
            Ph, HW, 0, vh + (size_t)b * sBC, HW, 0,
            ot + (size_t)b * sBC, CCH, 0, HW, nullptr, 1.f, 0, nullptr, 0);
    }

    // y[b,o,i] = (Ot @ wo^T)[i,o] + bo[o] + x[b,o,i]  M=4096 N=512 K=512
    gemm_wmma<<<dim3(CCH / BN, HW / BM, BATCH), 256, 0, stream>>>(
        ot, CCH, sBC, woh, CCH, 0, out, HW, sBC, CCH, bo, 1.f, 3, x, sBC);
}
